// EmotionTokenModule_83141976916851
// MI455X (gfx1250) — matrix operand order repaired and checked by reading, not compiled, not run on hardware
//
#include <hip/hip_runtime.h>
#include <math.h>

// ---------------------------------------------------------------------------
// VQ-VAE quantizer for MI455X (gfx1250, wave32, WMMA + TDM).
//
//   K0: scale codebook w = emb*|std|+mean, ||w||^2 per code, and write the
//       pre-swizzled split-BF16 (hi/lo) B-fragment image to global scratch.
//   K1: distance GEMM (3x v_wmma_f32_16x16x32_bf16 per 16x16 tile, split-BF16
//       error compensation) + argmin + per-pixel squared error + histogram.
//       Codebook halves staged into LDS via TENSOR_LOAD_TO_LDS (TDM), one
//       16KB 1-D descriptor per wave, s_wait_tensorcnt + barrier.
//   K2: cw = w @ conv_w^T + conv_b  (tiny: 256x512x512).
//   K3: out[b,o,h,w] = cw[idx[pixel]][o]  (gather + 134MB streaming write).
//   K4: finalize loss & perplexity (fixed-order reductions: deterministic).
// ---------------------------------------------------------------------------

typedef __bf16        bf16x16 __attribute__((ext_vector_type(16)));
typedef float         floatx8 __attribute__((ext_vector_type(8)));
typedef unsigned int  u32x4   __attribute__((ext_vector_type(4)));
typedef int           i32x4   __attribute__((ext_vector_type(4)));
typedef int           i32x8   __attribute__((ext_vector_type(8)));

#define HW 4096                 // H*W
#define CDIM 512
#define NCODE 256
#define NPIX 65536              // B*H*W
#define OUT_ELEMS 33554432ull   // B*C*H*W
#define FRAG_HALF_EL 131072     // bf16 elems per half (hi 65536 + lo 65536)

// Fragment position of codebook element (code, c) inside its half image.
// B layout per 32x16 tile: lane = n + 16*(k/16), in-lane t = k%16.
__device__ __forceinline__ int fragOffsetEl(int codeLocal, int c) {
    int ntile = codeLocal >> 4;
    int nIdx  = codeLocal & 15;
    int kc    = c >> 4;          // 16-wide k chunk
    int i     = c & 15;
    int kstep = kc >> 1;
    int lslot = nIdx + 16 * (kc & 1);
    return (((ntile * 16 + kstep) * 32) + lslot) * 16 + i;
}

// ------------------------------- K0 ----------------------------------------
__global__ void k0_prep(const float* __restrict__ emb, const float* __restrict__ stdp,
                        const float* __restrict__ means, float* __restrict__ w,
                        float* __restrict__ wsq, __bf16* __restrict__ fragG,
                        unsigned int* __restrict__ hist, float* __restrict__ lossPartial)
{
    __shared__ float red[256];
    const int j = blockIdx.x;        // code
    const int t = threadIdx.x;       // 256 threads
    const float sd = fabsf(stdp[0]);
    const float mn = (means[0] + means[1] + means[2]) * (1.0f / 3.0f);
    const int halfB = j >> 7;
    const int codeLocal = j & 127;
    __bf16* fh = fragG + halfB * FRAG_HALF_EL;          // hi region
    __bf16* fl = fh + 65536;                            // lo region
    float part = 0.0f;
    for (int c = t; c < CDIM; c += 256) {
        float x = emb[j * CDIM + c] * sd + mn;
        w[j * CDIM + c] = x;
        part += x * x;
        __bf16 h = (__bf16)x;
        __bf16 l = (__bf16)(x - (float)h);
        int off = fragOffsetEl(codeLocal, c);
        fh[off] = h;
        fl[off] = l;
    }
    red[t] = part;
    __syncthreads();
    if (t == 0) {
        float s = 0.0f;
        for (int i = 0; i < 256; ++i) s += red[i];   // fixed order: deterministic
        wsq[j] = s;
        hist[j] = 0u;
        lossPartial[j] = 0.0f;
    }
}

// ------------------------------- K1 ----------------------------------------
// Block: 512 threads (16 waves). Wave handles 16 pixels x all 256 codes.
// LDS: B-fragments for 128 codes (hi+lo bf16) = 256KB, + wsq + reduce scratch.
__global__ __launch_bounds__(512, 1) void k1_dist_argmin(
    const float* __restrict__ z, const __bf16* __restrict__ fragG,
    const float* __restrict__ wsq, int* __restrict__ idxOut,
    float* __restrict__ lossPartial, unsigned int* __restrict__ hist)
{
    extern __shared__ char smem[];
    __bf16* fragHi = (__bf16*)smem;                 // 131072 B
    __bf16* fragLo = (__bf16*)(smem + 131072);      // 131072 B
    float*  wsqS   = (float*)(smem + 262144);       // 1024 B
    float*  pmin   = (float*)(smem + 263168);       // 1024 B (per-pixel min dist)
    float*  pzsq   = (float*)(smem + 264192);       // 1024 B (per-pixel ||z||^2)

    const int tid    = threadIdx.x;
    const int lane   = tid & 31;
    const int waveId = tid >> 5;
    const int g      = lane >> 4;     // half-wave group
    const int li     = lane & 15;

    if (tid < NCODE) wsqS[tid] = wsq[tid];

    // This wave's 16 pixels; A-matrix lane holds row m = li (both groups).
    const int pixBase = blockIdx.x * 256 + waveId * 16;
    const int pix = pixBase + li;
    const int bb  = pix >> 12;            // batch (4096 px per batch, tiles aligned)
    const int hw  = pix & 4095;
    const float* zb = z + (size_t)bb * CDIM * HW + hw;   // + c*HW

    float minval[8];
    int   minidx[8];
#pragma unroll
    for (int v = 0; v < 8; ++v) { minval[v] = 3.4e38f; minidx[v] = 0; }
    float zsq = 0.0f;

    for (int half = 0; half < 2; ++half) {
        __syncthreads();  // previous compute done (and wsqS visible) before restage

        // ---- stage one 256KB codebook half (pre-swizzled hi+lo frags) ----
        const __bf16* src = fragG + half * FRAG_HALF_EL;
#if __has_builtin(__builtin_amdgcn_tensor_load_to_lds) && \
    __has_builtin(__builtin_amdgcn_s_wait_tensorcnt)
        {
            // One TDM descriptor per wave: 16KB contiguous slice (2048 x 8B).
            unsigned long long ga =
                (unsigned long long)(size_t)src + (unsigned long long)waveId * 16384ull;
            unsigned la = (unsigned)(size_t)(void*)smem + (unsigned)waveId * 16384u;
            u32x4 g0;
            g0[0] = 1u;                                  // count=1 (valid user D#)
            g0[1] = la;                                  // lds_addr
            g0[2] = (unsigned)ga;                        // global_addr[31:0]
            g0[3] = ((unsigned)(ga >> 32) & 0x01FFFFFFu) // global_addr[56:32]
                    | (2u << 30);                        // type = image
            i32x8 g1;
            g1[0] = (int)(3u << 16);        // workgroup_mask=0, data_size=8B
            g1[1] = (int)(2048u << 16);     // tensor_dim0 = 2048 (low 16 @ [63:48])
            g1[2] = (int)(1u << 16);        // tensor_dim1 = 1   (low 16 @ [95:80])
            g1[3] = (int)(2048u << 16);     // tile_dim0 = 2048  (@ [127:112])
            g1[4] = 0;                      // tile_dim1/2 unused
            g1[5] = 2048;                   // tensor_dim0_stride low32
            g1[6] = 0;
            g1[7] = 0;
            i32x4 g2 = {0, 0, 0, 0};        // VADDR2 group (unused, <=2D)
            i32x4 g3 = {0, 0, 0, 0};        // VADDR3 group (unused, <=2D)
            i32x8 g4 = {0, 0, 0, 0, 0, 0, 0, 0};  // trailing group (clang-23 form)
            __builtin_amdgcn_tensor_load_to_lds(g0, g1, g2, g3, g4, 0);
            __builtin_amdgcn_s_wait_tensorcnt(0);
        }
#else
        {
            // Fallback: plain vector copy (host pass / toolchains without TDM).
            const uint4* s4 = (const uint4*)src;
            uint4* d4 = (uint4*)smem;
#pragma unroll
            for (int it = 0; it < 32; ++it) {
                int e = tid + it * 512;      // 16384 x 16B = 256KB
                d4[e] = s4[e];
            }
        }
#endif
        __syncthreads();

        floatx8 acc[8];
#pragma unroll
        for (int nt = 0; nt < 8; ++nt) acc[nt] = (floatx8){0, 0, 0, 0, 0, 0, 0, 0};

        for (int kstep = 0; kstep < 16; ++kstep) {
            // ---- build A hi/lo fragments (contiguous k = 16g + s, same order as B image) ----
            // lane group g: VGPR0-3 -> k = base+16g+{0..7}, VGPR4-7 -> +8.
            bf16x16 aHi, aLo;
            const int c0 = kstep * 32 + 16 * g;
#pragma unroll
            for (int j = 0; j < 8; ++j) {
                float x = zb[(size_t)(c0 + j) * HW];
                __bf16 h = (__bf16)x;
                aHi[j] = h; aLo[j] = (__bf16)(x - (float)h);
                if (half == 0) zsq += x * x;
            }
            const int c1 = c0 + 8;
#pragma unroll
            for (int j = 0; j < 8; ++j) {
                float x = zb[(size_t)(c1 + j) * HW];
                __bf16 h = (__bf16)x;
                aHi[8 + j] = h; aLo[8 + j] = (__bf16)(x - (float)h);
                if (half == 0) zsq += x * x;
            }
            // ---- 8 code tiles x error-compensated split-BF16 (3 WMMAs) ----
#pragma unroll
            for (int nt = 0; nt < 8; ++nt) {
                int boff = (((nt * 16 + kstep) * 32) + lane) * 16;
                bf16x16 bHi = *(bf16x16*)(fragHi + boff);
                bf16x16 bLo = *(bf16x16*)(fragLo + boff);
                acc[nt] = __builtin_amdgcn_wmma_f32_16x16x32_bf16(
                    false, aHi, false, bHi, (short)0, acc[nt], false, false);
                acc[nt] = __builtin_amdgcn_wmma_f32_16x16x32_bf16(
                    false, aHi, false, bLo, (short)0, acc[nt], false, false);
                acc[nt] = __builtin_amdgcn_wmma_f32_16x16x32_bf16(
                    false, aLo, false, bHi, (short)0, acc[nt], false, false);
            }
        }

        // ---- fold into running argmin. D layout: lane holds col n = li,
        //      rows m = v + 8*g across the 8 accumulator VGPRs. ----
#pragma unroll
        for (int nt = 0; nt < 8; ++nt) {
            int n = half * 128 + nt * 16 + li;
            float wq = wsqS[n];
#pragma unroll
            for (int v = 0; v < 8; ++v) {
                float d = wq - 2.0f * acc[nt][v];   // ||w||^2 - 2 z.w
                if (d < minval[v]) { minval[v] = d; minidx[v] = n; }
            }
        }
    }

    // ---- cross-lane argmin over the 16 columns in each half-wave group ----
#pragma unroll
    for (int v = 0; v < 8; ++v) {
        for (int m = 1; m < 16; m <<= 1) {
            float ov = __shfl_xor(minval[v], m, 32);
            int   oi = __shfl_xor(minidx[v], m, 32);
            if (ov < minval[v] || (ov == minval[v] && oi < minidx[v])) {
                minval[v] = ov; minidx[v] = oi;
            }
        }
    }
    zsq += __shfl_xor(zsq, 16, 32);   // combine the two k-halves per pixel

    if (li == 0) {
#pragma unroll
        for (int v = 0; v < 8; ++v) {
            int m = v + 8 * g;
            int n = pixBase + m;
            idxOut[n] = minidx[v];
            pmin[waveId * 16 + m] = minval[v];
            atomicAdd(&hist[minidx[v]], 1u);   // integer atomics: deterministic
        }
    }
    if (g == 0) pzsq[waveId * 16 + li] = zsq;
    __syncthreads();
    if (tid < 256) pmin[tid] += pzsq[tid];     // ||z-w||^2 = ||z||^2 + min_d
    __syncthreads();
    if (tid == 0) {
        float s = 0.0f;
        for (int i = 0; i < 256; ++i) s += pmin[i];   // fixed order
        lossPartial[blockIdx.x] = s;
    }
}

// ------------------------------- K2 ----------------------------------------
__global__ void k2_conv_codebook(const float* __restrict__ w, const float* __restrict__ convW,
                                 const float* __restrict__ convB, float* __restrict__ cw)
{
    __shared__ float wrow[CDIM];
    const int j = blockIdx.x;     // code
    const int t = threadIdx.x;    // 256
    wrow[t]       = w[j * CDIM + t];
    wrow[t + 256] = w[j * CDIM + t + 256];
    __syncthreads();
#pragma unroll
    for (int oo = 0; oo < 2; ++oo) {
        int o = t + oo * 256;
        const float* cr = convW + (size_t)o * CDIM;
        float acc = convB[o];
#pragma unroll 4
        for (int c = 0; c < CDIM; ++c) acc = fmaf(wrow[c], cr[c], acc);
        cw[j * CDIM + o] = acc;
    }
}

// ------------------------------- K3 ----------------------------------------
__global__ __launch_bounds__(256) void k3_gather_out(
    const int* __restrict__ idx, const float* __restrict__ cw, float* __restrict__ out)
{
    const int t = threadIdx.x;
    const int px = blockIdx.x * 64 + (t & 63);   // 1024 blocks x 64 pixels
    const int oBase = (t >> 6) * 128;            // 4 channel strips
    const int code = idx[px];
    const float* cr = cw + (size_t)code * CDIM + oBase;
    const int b  = px >> 12;
    const int hw = px & 4095;
    float* ob = out + ((size_t)b * CDIM + oBase) * HW + hw;
#pragma unroll 8
    for (int o = 0; o < 128; ++o) ob[(size_t)o * HW] = cr[o];
}

// ------------------------------- K4 ----------------------------------------
__global__ void k4_finalize(const unsigned int* __restrict__ hist,
                            const float* __restrict__ lossPartial,
                            float* __restrict__ outTail)
{
    __shared__ float ent[256];
    __shared__ float lp[256];
    const int t = threadIdx.x;
    float e = (float)hist[t] * (1.0f / (float)NPIX);
    ent[t] = e * logf(e + 1e-10f);
    lp[t]  = lossPartial[t];
    __syncthreads();
    if (t == 0) {
        float es = 0.0f, ls = 0.0f;
        for (int i = 0; i < 256; ++i) { es += ent[i]; ls += lp[i]; }
        outTail[0] = 1.25f * ls / (float)OUT_ELEMS;  // (1+BETA)*MSE
        outTail[1] = expf(-es);                      // perplexity
    }
}

// ---------------------------------------------------------------------------
extern "C" void kernel_launch(void* const* d_in, const int* in_sizes, int n_in,
                              void* d_out, int out_size, void* d_ws, size_t ws_size,
                              hipStream_t stream)
{
    const float* z     = (const float*)d_in[0];
    const float* emb   = (const float*)d_in[1];
    const float* stdp  = (const float*)d_in[2];
    const float* means = (const float*)d_in[3];
    const float* convW = (const float*)d_in[4];
    const float* convB = (const float*)d_in[5];
    float* out = (float*)d_out;

    // Workspace layout (floats): w[131072] wsq[256] cw[131072] hist[256]
    //                            lossPartial[256] idx[65536] fragG[bf16 x 262144]
    float* w            = (float*)d_ws;
    float* wsq          = w + 131072;
    float* cw           = wsq + 256;
    unsigned int* hist  = (unsigned int*)(cw + 131072);
    float* lossPartial  = (float*)(hist + 256);
    int*   idx          = (int*)(lossPartial + 256);
    __bf16* fragG       = (__bf16*)(idx + NPIX);   // 524288 B, 8B-aligned

    // Idempotent host-side attribute set (no static guards — harness rule).
    (void)hipFuncSetAttribute((const void*)k1_dist_argmin,
                              hipFuncAttributeMaxDynamicSharedMemorySize, 265216);

    k0_prep<<<256, 256, 0, stream>>>(emb, stdp, means, w, wsq, fragG, hist, lossPartial);
    k1_dist_argmin<<<256, 512, 265216, stream>>>(z, fragG, wsq, idx, lossPartial, hist);
    k2_conv_codebook<<<256, 256, 0, stream>>>(w, convW, convB, cw);
    k3_gather_out<<<1024, 256, 0, stream>>>(idx, cw, out);
    k4_finalize<<<1, 256, 0, stream>>>(hist, lossPartial, out + OUT_ELEMS);
}
